// LinearAttentionBlock_58884001628510
// MI455X (gfx1250) — matrix operand order repaired and checked by reading, not compiled, not run on hardware
//
#include <hip/hip_runtime.h>
#include <hip/hip_bf16.h>

typedef __attribute__((ext_vector_type(16))) __bf16 v16bf;
typedef __attribute__((ext_vector_type(8)))  float  v8f;

#define BATCH 8
#define CH    128
#define SEQ   16384
#define NGRP  8

union ABFrag { v16bf v; uint4 q[2]; __bf16 h[16]; };
union BF8    { uint4 q; __bf16 h[8]; };
union BF4    { unsigned long long u; __bf16 h[4]; };

static __device__ __forceinline__ v8f wmma_bf16(const ABFrag& a, const ABFrag& b, v8f c) {
  return __builtin_amdgcn_wmma_f32_16x16x32_bf16(false, a.v, false, b.v, (short)0, c, false, false);
}

static __device__ __forceinline__ v8f zero8() {
  v8f z = {0.f,0.f,0.f,0.f,0.f,0.f,0.f,0.f};
  return z;
}

// ---------------- K1a: fp32 x -> bf16 Xb ----------------
__global__ __launch_bounds__(256) void cvt_x_kernel(const float* __restrict__ x,
                                                    __bf16* __restrict__ xb) {
  size_t idx = (size_t)blockIdx.x * 256 + threadIdx.x;
  size_t base = idx * 4;
  float4 f = *(const float4*)(x + base);
  BF4 o;
  o.h[0] = (__bf16)f.x; o.h[1] = (__bf16)f.y;
  o.h[2] = (__bf16)f.z; o.h[3] = (__bf16)f.w;
  *(unsigned long long*)(xb + base) = o.u;
}

// ---------------- K1b: fp32 Wq/Wk/Wv -> bf16 [3][C*C] ----------------
__global__ __launch_bounds__(256) void cvt_w_kernel(const float* __restrict__ wq,
                                                    const float* __restrict__ wk,
                                                    const float* __restrict__ wv,
                                                    __bf16* __restrict__ wb) {
  int i = blockIdx.x * 256 + threadIdx.x;
  if (i >= 3 * CH * CH) return;
  float v;
  if (i < CH * CH)              v = wq[i];
  else if (i < 2 * CH * CH)     v = wk[i - CH * CH];
  else                          v = wv[i - 2 * CH * CH];
  wb[i] = (__bf16)v;
}

// ---------------- K2: Q/K/V projection: P^T = W_p @ X + b_p  ([C][S] bf16 out) ----------------
__global__ __launch_bounds__(256) void qkv_gemm_kernel(const __bf16* __restrict__ Xb,
                                                       const __bf16* __restrict__ Wb,
                                                       const float* __restrict__ bq,
                                                       const float* __restrict__ bk,
                                                       const float* __restrict__ bv,
                                                       __bf16* __restrict__ Qt,
                                                       __bf16* __restrict__ Kt,
                                                       __bf16* __restrict__ Vt) {
  __shared__ __bf16 XT[64][136];   // [s_local][c], padded stride
  const int b  = blockIdx.y;
  const int s0 = blockIdx.x * 64;
  const int tid = threadIdx.x;

  // stage X block [c:0..128) x [s0..s0+64) transposed into LDS
  const __bf16* xsrc = Xb + (size_t)b * CH * SEQ;
  #pragma unroll
  for (int i = 0; i < 4; i++) {
    int u = tid + 256 * i;          // 0..1023
    int c  = u >> 3;                // 0..127
    int s8 = (u & 7) * 8;
    BF8 v; v.q = *(const uint4*)(xsrc + (size_t)c * SEQ + s0 + s8);
    #pragma unroll
    for (int e = 0; e < 8; e++) XT[s8 + e][c] = v.h[e];
  }
  __syncthreads();

  const int w    = tid >> 5;   // wave id == d-block (group of 16 output rows)
  const int lane = tid & 31;
  const int r    = lane & 15;
  const int half = lane >> 4;

  for (int p = 0; p < 3; p++) {
    const __bf16* Wrow = Wb + (size_t)p * CH * CH + (size_t)(16 * w + r) * CH;
    const float*  bias = (p == 0) ? bq : ((p == 1) ? bk : bv);
    __bf16*       Op   = ((p == 0) ? Qt : ((p == 1) ? Kt : Vt)) + (size_t)b * CH * SEQ;

    ABFrag a[4];
    #pragma unroll
    for (int k4 = 0; k4 < 4; k4++) {
      int k0 = 32 * k4;
      a[k4].q[0] = *(const uint4*)(Wrow + k0 + half * 8);
      a[k4].q[1] = *(const uint4*)(Wrow + k0 + 16 + half * 8);
    }

    #pragma unroll
    for (int j = 0; j < 4; j++) {
      v8f acc = zero8();
      int sl = 16 * j + r;
      #pragma unroll
      for (int k4 = 0; k4 < 4; k4++) {
        int k0 = 32 * k4;
        ABFrag bb;
        bb.q[0] = *(const uint4*)(&XT[sl][k0 + half * 8]);
        bb.q[1] = *(const uint4*)(&XT[sl][k0 + 16 + half * 8]);
        acc = wmma_bf16(a[k4], bb, acc);
      }
      int scol = s0 + 16 * j + r;
      #pragma unroll
      for (int jv = 0; jv < 8; jv++) {
        int d = 16 * w + jv + 8 * half;
        Op[(size_t)d * SEQ + scol] = (__bf16)(acc[jv] + bias[d]);
      }
    }
  }
}

// ---------------- K3: softmax over S for each (b,c) row of Qt and Kt (in place) ----------------
__global__ __launch_bounds__(256) void softmax_seq_kernel(__bf16* Qt, __bf16* Kt) {
  __shared__ float red[256];
  const int c = blockIdx.x, b = blockIdx.y;
  __bf16* base = (blockIdx.z == 0 ? Qt : Kt) + ((size_t)b * CH + c) * SEQ;
  const int tid = threadIdx.x;

  float m = -1e30f;
  for (int i = tid; i < SEQ; i += 256) m = fmaxf(m, (float)base[i]);
  red[tid] = m; __syncthreads();
  for (int o = 128; o > 0; o >>= 1) { if (tid < o) red[tid] = fmaxf(red[tid], red[tid + o]); __syncthreads(); }
  float gm = red[0]; __syncthreads();

  float sum = 0.f;
  for (int i = tid; i < SEQ; i += 256) sum += __expf((float)base[i] - gm);
  red[tid] = sum; __syncthreads();
  for (int o = 128; o > 0; o >>= 1) { if (tid < o) red[tid] += red[tid + o]; __syncthreads(); }
  float inv = 1.0f / red[0];

  for (int i = tid; i < SEQ; i += 256) base[i] = (__bf16)(__expf((float)base[i] - gm) * inv);
}

// ---------------- K4: KVT[d][c] = sum_s V^T[d,s] * K^T[c,s]  (fp32 atomic accumulate) ----------------
__global__ __launch_bounds__(256) void kv_gemm_kernel(const __bf16* __restrict__ Kt,
                                                      const __bf16* __restrict__ Vt,
                                                      float* __restrict__ KVT) {
  const int b  = blockIdx.y;
  const int sc = blockIdx.x;           // 16 chunks of 1024 along S
  const int tid = threadIdx.x;
  const int w = tid >> 5, lane = tid & 31, r = lane & 15, half = lane >> 4;

  const __bf16* Arow  = Vt + (size_t)b * CH * SEQ + (size_t)(16 * w + r) * SEQ + sc * 1024;
  const __bf16* Bbase = Kt + (size_t)b * CH * SEQ + sc * 1024;

  v8f acc[8];
  #pragma unroll
  for (int n = 0; n < 8; n++) acc[n] = zero8();

  for (int ks = 0; ks < 32; ks++) {
    int k0 = ks * 32;
    ABFrag a;
    a.q[0] = *(const uint4*)(Arow + k0 + half * 8);
    a.q[1] = *(const uint4*)(Arow + k0 + 16 + half * 8);
    #pragma unroll
    for (int n = 0; n < 8; n++) {
      const __bf16* Brow = Bbase + (size_t)(16 * n + r) * SEQ;
      ABFrag bb;
      bb.q[0] = *(const uint4*)(Brow + k0 + half * 8);
      bb.q[1] = *(const uint4*)(Brow + k0 + 16 + half * 8);
      acc[n] = wmma_bf16(a, bb, acc[n]);
    }
  }

  float* outb = KVT + (size_t)b * CH * CH;
  #pragma unroll
  for (int n = 0; n < 8; n++) {
    #pragma unroll
    for (int jv = 0; jv < 8; jv++) {
      int d  = 16 * w + jv + 8 * half;
      int cc = 16 * n + r;
      atomicAdd(&outb[d * CH + cc], acc[n][jv]);
    }
  }
}

// ---------------- K5: Y[d][s] = x[d][s] + sum_c KVT[d][c]*Qsm[c][s]; fused GroupNorm partial sums ----------------
__global__ __launch_bounds__(256) void z_gemm_kernel(const __bf16* __restrict__ Qt,
                                                     const float* __restrict__ KVT,
                                                     const float* __restrict__ x,
                                                     float* __restrict__ Y,
                                                     float* __restrict__ stats) {
  __shared__ __bf16 XT[64][136];
  const int b  = blockIdx.y;
  const int s0 = blockIdx.x * 64;
  const int tid = threadIdx.x;

  const __bf16* qsrc = Qt + (size_t)b * CH * SEQ;
  #pragma unroll
  for (int i = 0; i < 4; i++) {
    int u = tid + 256 * i;
    int c  = u >> 3;
    int s8 = (u & 7) * 8;
    BF8 v; v.q = *(const uint4*)(qsrc + (size_t)c * SEQ + s0 + s8);
    #pragma unroll
    for (int e = 0; e < 8; e++) XT[s8 + e][c] = v.h[e];
  }
  __syncthreads();

  const int w = tid >> 5, lane = tid & 31, r = lane & 15, half = lane >> 4;

  // A = KVT rows (fp32 -> bf16 on load)
  const float* Ar = KVT + (size_t)b * CH * CH + (size_t)(16 * w + r) * CH;
  ABFrag a[4];
  #pragma unroll
  for (int k4 = 0; k4 < 4; k4++) {
    int k0 = 32 * k4;
    #pragma unroll
    for (int e = 0; e < 8; e++) {
      a[k4].h[e]     = (__bf16)Ar[k0 + half * 8 + e];
      a[k4].h[8 + e] = (__bf16)Ar[k0 + 16 + half * 8 + e];
    }
  }

  const float* xb = x + (size_t)b * CH * SEQ;
  float*       yb = Y + (size_t)b * CH * SEQ;
  float sum = 0.f, ss = 0.f;

  #pragma unroll
  for (int j = 0; j < 4; j++) {
    v8f acc = zero8();
    int sl = 16 * j + r;
    #pragma unroll
    for (int k4 = 0; k4 < 4; k4++) {
      int k0 = 32 * k4;
      ABFrag bb;
      bb.q[0] = *(const uint4*)(&XT[sl][k0 + half * 8]);
      bb.q[1] = *(const uint4*)(&XT[sl][k0 + 16 + half * 8])
      acc = wmma_bf16(a[k4], bb, acc);
    }
    int scol = s0 + 16 * j + r;
    #pragma unroll
    for (int jv = 0; jv < 8; jv++) {
      int d = 16 * w + jv + 8 * half;
      float y = acc[jv] + xb[(size_t)d * SEQ + scol];
      yb[(size_t)d * SEQ + scol] = y;
      sum += y; ss += y * y;
    }
  }

  // wave tile rows span exactly group g == w; reduce across 32 lanes
  for (int o = 16; o > 0; o >>= 1) { sum += __shfl_xor(sum, o, 32); ss += __shfl_xor(ss, o, 32); }
  if (lane == 0) {
    atomicAdd(&stats[(b * NGRP + w) * 2 + 0], sum);
    atomicAdd(&stats[(b * NGRP + w) * 2 + 1], ss);
  }
}

// ---------------- K6: GroupNorm finalize + [C,S] -> [S,C] permuted output ----------------
__global__ __launch_bounds__(256) void finalize_kernel(const float* __restrict__ Y,
                                                       const float* __restrict__ stats,
                                                       const float* __restrict__ gw,
                                                       const float* __restrict__ gb,
                                                       float* __restrict__ out) {
  __shared__ float T[64][33];
  const int b  = blockIdx.z;
  const int c0 = blockIdx.y * 32;
  const int s0 = blockIdx.x * 64;
  const int tid = threadIdx.x;

  const float inv_cnt = 1.0f / ((CH / NGRP) * (float)SEQ); // 1/262144
  int cl = tid >> 3;          // 0..31
  int sb = (tid & 7) * 8;     // 0..56
  int c  = c0 + cl;
  int g  = c >> 4;
  float mean = stats[(b * NGRP + g) * 2 + 0] * inv_cnt;
  float var  = stats[(b * NGRP + g) * 2 + 1] * inv_cnt - mean * mean;
  float rstd = rsqrtf(var + 1e-5f);
  float sc = rstd * gw[c];
  float sh = gb[c] - mean * sc;

  const float* yr = Y + (size_t)b * CH * SEQ + (size_t)c * SEQ + s0 + sb;
  #pragma unroll
  for (int e = 0; e < 8; e++) T[sb + e][cl] = yr[e] * sc + sh;
  __syncthreads();

  int cl2 = tid & 31;
  int sst = (tid >> 5) * 8;
  float* ob = out + (size_t)b * SEQ * CH + (size_t)s0 * CH + c0 + cl2;
  #pragma unroll
  for (int e = 0; e < 8; e++) ob[(size_t)(sst + e) * CH] = T[sst + e][cl2];
}

extern "C" void kernel_launch(void* const* d_in, const int* in_sizes, int n_in,
                              void* d_out, int out_size, void* d_ws, size_t ws_size,
                              hipStream_t stream) {
  const float* x  = (const float*)d_in[0];
  const float* Wq = (const float*)d_in[1];
  const float* bq = (const float*)d_in[2];
  const float* Wk = (const float*)d_in[3];
  const float* bk = (const float*)d_in[4];
  const float* Wv = (const float*)d_in[5];
  const float* bv = (const float*)d_in[6];
  const float* gw = (const float*)d_in[7];
  const float* gb = (const float*)d_in[8];
  float* out = (float*)d_out;

  const size_t NB = (size_t)BATCH * CH * SEQ;   // 16,777,216 elements
  char* ws = (char*)d_ws;
  size_t off = 0;
  auto carve = [&](size_t bytes) -> void* {
    void* p = ws + off;
    off += (bytes + 255) & ~(size_t)255;
    return p;
  };
  __bf16* Xb   = (__bf16*)carve(NB * 2);                    // bf16 x, [B][C][S]
  __bf16* Wb   = (__bf16*)carve((size_t)3 * CH * CH * 2);   // bf16 weights
  __bf16* Qt   = (__bf16*)carve(NB * 2);
  __bf16* Kt   = (__bf16*)carve(NB * 2);
  __bf16* Vt   = (__bf16*)carve(NB * 2);
  float*  KVT  = (float*)carve((size_t)BATCH * CH * CH * 4);
  float*  Yf   = (float*)carve(NB * 4);
  float*  stats = (float*)carve((size_t)BATCH * NGRP * 2 * 4);

  dim3 blk(256);
  cvt_x_kernel<<<(unsigned)(NB / (256 * 4)), blk, 0, stream>>>(x, Xb);
  cvt_w_kernel<<<(3 * CH * CH + 255) / 256, blk, 0, stream>>>(Wq, Wk, Wv, Wb);
  qkv_gemm_kernel<<<dim3(SEQ / 64, BATCH), blk, 0, stream>>>(Xb, Wb, bq, bk, bv, Qt, Kt, Vt);
  softmax_seq_kernel<<<dim3(CH, BATCH, 2), blk, 0, stream>>>(Qt, Kt);
  hipMemsetAsync(KVT, 0, (size_t)BATCH * CH * CH * 4, stream);
  hipMemsetAsync(stats, 0, (size_t)BATCH * NGRP * 2 * 4, stream);
  kv_gemm_kernel<<<dim3(16, BATCH), blk, 0, stream>>>(Kt, Vt, KVT);
  z_gemm_kernel<<<dim3(SEQ / 64, BATCH), blk, 0, stream>>>(Qt, KVT, x, Yf, stats);
  finalize_kernel<<<dim3(SEQ / 64, CH / 32, BATCH), blk, 0, stream>>>(Yf, stats, gw, gb, out);
}
